// MultiHeadAttention_30605936951344
// MI455X (gfx1250) — matrix operand order repaired and checked by reading, not compiled, not run on hardware
//
#include <hip/hip_runtime.h>
#include <math.h>

// ---------------------------------------------------------------------------
// MHA on MI455X (gfx1250): bf16 WMMA pipeline, round 2.
//  - GEMMs register-blocked 16x64 per wave (1 A-frag reused by 4 B-frags)
//  - 4 waves / workgroup on adjacent row-tiles -> structural L0 reuse of
//    B-operands (GEMMs) and K/V tiles (attention); no inter-wave sync needed
//  - flash attention: QK^T wmma -> online softmax (shfl_xor reductions) ->
//    P relayout through per-wave LDS slab -> P*V wmma
// All WMMA operand loads are contiguous b128 loads via pre-transposed B
// operands (A-/B-frag lane layouts per CDNA5 ISA 7.12.2).
// ---------------------------------------------------------------------------

#define S_LEN 4096
#define D_MOD 1024
#define NHEAD 16
#define HDIM  64

typedef __attribute__((ext_vector_type(16))) __bf16 v16bf;
typedef __attribute__((ext_vector_type(8)))  float  v8f;

struct alignas(16) U128 { unsigned int x, y, z, w; };
struct FragU { U128 lo, hi; };

__device__ __forceinline__ unsigned short f2bf(float x) {
    unsigned int u = __builtin_bit_cast(unsigned int, x);
    unsigned int r = (u + 0x7FFFu + ((u >> 16) & 1u)) >> 16;
    return (unsigned short)r;
}

// A-operand frag (16x32 bf16, M x K): lane m holds row m, two 16B chunks at
// K = 8*half and K = 16 + 8*half.  base is row-major [.., ld] bf16.
__device__ __forceinline__ v16bf load_a(const unsigned short* base, int ld,
                                        int m0, int k0) {
    const int lane = threadIdx.x & 31;
    const int m = lane & 15, hf = lane >> 4;
    const unsigned short* p = base + (size_t)(m0 + m) * ld + k0;
    FragU f;
    f.lo = *(const U128*)(p + 8 * hf);
    f.hi = *(const U128*)(p + 16 + 8 * hf);
    return __builtin_bit_cast(v16bf, f);
}

// B-operand frag (32x16 bf16, K x N): lane n holds column n; two 16B chunks at
// K = 8*half and K = 16 + 8*half.  base is row-major B^T
// (rows = N, ld along K).
__device__ __forceinline__ v16bf load_b(const unsigned short* base, int ld,
                                        int n0, int k0) {
    const int lane = threadIdx.x & 31;
    const int nn = lane & 15, hf = lane >> 4;
    const unsigned short* p = base + (size_t)(n0 + nn) * ld + k0;
    FragU f;
    f.lo = *(const U128*)(p + 8 * hf);
    f.hi = *(const U128*)(p + 16 + 8 * hf);
    return __builtin_bit_cast(v16bf, f);
}

__device__ __forceinline__ v8f wmma_bf16(v16bf a, v16bf b, v8f c) {
    return __builtin_amdgcn_wmma_f32_16x16x32_bf16(
        false, a, false, b, (short)0, c, false, false);
}

// ---------------------------------------------------------------------------
// Prep kernels
// ---------------------------------------------------------------------------
__global__ void k_f32_to_bf16(const float* __restrict__ src,
                              unsigned short* __restrict__ dst, int n) {
    int i = blockIdx.x * blockDim.x + threadIdx.x;
    if (i < n) dst[i] = f2bf(src[i]);
}

// W: [H, D, E] f32 -> WT: [H, E, D] bf16
__global__ void k_transpose_w(const float* __restrict__ W,
                              unsigned short* __restrict__ WT) {
    int i = blockIdx.x * blockDim.x + threadIdx.x;
    if (i >= NHEAD * HDIM * D_MOD) return;
    int h = i / (HDIM * D_MOD);
    int rem = i % (HDIM * D_MOD);
    int e = rem / D_MOD;
    int d = rem % D_MOD;
    WT[i] = f2bf(W[((size_t)h * D_MOD + d) * HDIM + e]);
}

// Wo: [HE, D] f32 -> WoT: [D, HE] bf16   (rows of B^T for the out-proj GEMM)
__global__ void k_transpose_wo(const float* __restrict__ Wo,
                               unsigned short* __restrict__ WoT) {
    int i = blockIdx.x * blockDim.x + threadIdx.x;
    if (i >= D_MOD * D_MOD) return;
    int nn = i / D_MOD;
    int kk = i % D_MOD;
    WoT[i] = f2bf(Wo[(size_t)kk * D_MOD + nn]);
}

// ---------------------------------------------------------------------------
// QKV projection: each wave computes a 16x64 tile (full E) for one matrix;
// one A-frag feeds 4 wmmas per K-step.  4 waves/block on adjacent s-tiles.
// grid: (S/64, H, 3).  Q,K stored [H,S,E]; V stored transposed [H,E,S].
// ---------------------------------------------------------------------------
__global__ __launch_bounds__(128) void k_qkv(
    const unsigned short* __restrict__ Xb,
    const unsigned short* __restrict__ WqT,
    const unsigned short* __restrict__ WkT,
    const unsigned short* __restrict__ WvT,
    const float* __restrict__ bq, const float* __restrict__ bk,
    const float* __restrict__ bv,
    unsigned short* __restrict__ Qb, unsigned short* __restrict__ Kb,
    unsigned short* __restrict__ Vt) {
    const int w = threadIdx.x >> 5;
    const int s0 = (blockIdx.x * 4 + w) * 16;
    const int h = blockIdx.y;
    const int mat = blockIdx.z;
    const unsigned short* WT = (mat == 0) ? WqT : (mat == 1) ? WkT : WvT;
    const float* bias = (mat == 0) ? bq : (mat == 1) ? bk : bv;
    const unsigned short* WTh = WT + (size_t)h * HDIM * D_MOD;
    const int lane = threadIdx.x & 31, nn = lane & 15, hf = lane >> 4;

    v8f acc[4] = {};
    for (int k0 = 0; k0 < D_MOD; k0 += 32) {
        v16bf a = load_a(Xb, D_MOD, s0, k0);
#pragma unroll
        for (int j = 0; j < 4; ++j)
            acc[j] = wmma_bf16(a, load_b(WTh, D_MOD, j * 16, k0), acc[j]);
    }
#pragma unroll
    for (int j = 0; j < 4; ++j) {
        const float bb = bias[h * HDIM + j * 16 + nn];
#pragma unroll
        for (int r = 0; r < 8; ++r) {
            const int mr = r + hf * 8;
            unsigned short ob = f2bf(acc[j][r] + bb);
            if (mat == 0)
                Qb[((size_t)h * S_LEN + s0 + mr) * HDIM + j * 16 + nn] = ob;
            else if (mat == 1)
                Kb[((size_t)h * S_LEN + s0 + mr) * HDIM + j * 16 + nn] = ob;
            else
                Vt[((size_t)h * HDIM + j * 16 + nn) * S_LEN + s0 + mr] = ob;
        }
    }
}

// ---------------------------------------------------------------------------
// Flash attention: 4 waves/block, each on its own 16-query tile of the same
// head; identical K/V frag addresses across waves -> WGP L0 reuse.
// ---------------------------------------------------------------------------
__global__ __launch_bounds__(128) void k_attn(
    const unsigned short* __restrict__ Qb, const unsigned short* __restrict__ Kb,
    const unsigned short* __restrict__ Vt, const float* __restrict__ scale,
    unsigned short* __restrict__ Obf) {
    const int w = threadIdx.x >> 5;
    const int s0 = (blockIdx.x * 4 + w) * 16;
    const int h = blockIdx.y;
    const int lane = threadIdx.x & 31, nn = lane & 15, hf = lane >> 4;
    const unsigned short* Qh = Qb + (size_t)h * S_LEN * HDIM;
    const unsigned short* Kh = Kb + (size_t)h * S_LEN * HDIM;
    const unsigned short* Vh = Vt + (size_t)h * HDIM * S_LEN;
    const float sc = scale[h];

    const v16bf qa0 = load_a(Qh, HDIM, s0, 0);
    const v16bf qa1 = load_a(Qh, HDIM, s0, 32);

    float m[8], l[8];
    v8f acc[4] = {};
#pragma unroll
    for (int r = 0; r < 8; ++r) { m[r] = -3.0e38f; l[r] = 0.0f; }

    __shared__ unsigned short Pall[4][16 * 32];
    unsigned short* P = Pall[w];

    for (int k0 = 0; k0 < S_LEN; k0 += 32) {
        // ---- scores: 16 queries x 32 keys (C layout, f32) ----
        v8f c0 = {}, c1 = {};
        c0 = wmma_bf16(qa0, load_b(Kh, HDIM, k0, 0), c0);
        c0 = wmma_bf16(qa1, load_b(Kh, HDIM, k0, 32), c0);
        c1 = wmma_bf16(qa0, load_b(Kh, HDIM, k0 + 16, 0), c1);
        c1 = wmma_bf16(qa1, load_b(Kh, HDIM, k0 + 16, 32), c1);
        __builtin_prefetch(Kh + (size_t)(k0 + 32) * HDIM, 0, 0);

        // ---- online softmax, write P (bf16) to LDS row-major ----
#pragma unroll
        for (int r = 0; r < 8; ++r) {
            float x0 = c0[r] * sc, x1 = c1[r] * sc;
            float mx = fmaxf(x0, x1);
            mx = fmaxf(mx, __shfl_xor(mx, 1, 32));
            mx = fmaxf(mx, __shfl_xor(mx, 2, 32));
            mx = fmaxf(mx, __shfl_xor(mx, 4, 32));
            mx = fmaxf(mx, __shfl_xor(mx, 8, 32));
            float mn = fmaxf(m[r], mx);
            float alpha = __expf(m[r] - mn);
            m[r] = mn;
            float p0 = __expf(x0 - mn), p1 = __expf(x1 - mn);
            float rs = p0 + p1;
            rs += __shfl_xor(rs, 1, 32);
            rs += __shfl_xor(rs, 2, 32);
            rs += __shfl_xor(rs, 4, 32);
            rs += __shfl_xor(rs, 8, 32);
            l[r] = l[r] * alpha + rs;
#pragma unroll
            for (int j = 0; j < 4; ++j) acc[j][r] *= alpha;
            const int mr = r + hf * 8;
            P[mr * 32 + nn] = f2bf(p0);
            P[mr * 32 + nn + 16] = f2bf(p1);
        }

        // ---- P (A layout via LDS) * V (B frags from V^T rows) ----
        v16bf pa = load_a((const unsigned short*)P, 32, 0, 0);
#pragma unroll
        for (int j = 0; j < 4; ++j)
            acc[j] = wmma_bf16(pa, load_b(Vh, S_LEN, j * 16, k0), acc[j]);
    }

    // epilogue: normalize, write concat-head bf16 O [S, H*E]
#pragma unroll
    for (int j = 0; j < 4; ++j) {
#pragma unroll
        for (int r = 0; r < 8; ++r) {
            const int mr = r + hf * 8;
            float o = acc[j][r] / l[r];
            Obf[(size_t)(s0 + mr) * (NHEAD * HDIM) + h * HDIM + j * 16 + nn] =
                f2bf(o);
        }
    }
}

// ---------------------------------------------------------------------------
// Output projection: out = O * Wo + bo (f32 out).  16x64 tile per wave,
// 4 waves/block on adjacent s-tiles.  grid: (S/64, D/64).
// ---------------------------------------------------------------------------
__global__ __launch_bounds__(128) void k_out(
    const unsigned short* __restrict__ Obf,
    const unsigned short* __restrict__ WoT,
    const float* __restrict__ bo, float* __restrict__ out) {
    const int w = threadIdx.x >> 5;
    const int s0 = (blockIdx.x * 4 + w) * 16;
    const int n0 = blockIdx.y * 64;
    const int lane = threadIdx.x & 31, nn = lane & 15, hf = lane >> 4;

    v8f acc[4] = {};
    for (int k0 = 0; k0 < D_MOD; k0 += 32) {
        v16bf a = load_a(Obf, D_MOD, s0, k0);
#pragma unroll
        for (int j = 0; j < 4; ++j)
            acc[j] = wmma_bf16(a, load_b(WoT, D_MOD, n0 + j * 16, k0), acc[j]);
    }
#pragma unroll
    for (int j = 0; j < 4; ++j) {
        const float bb = bo[n0 + j * 16 + nn];
#pragma unroll
        for (int r = 0; r < 8; ++r)
            out[(size_t)(s0 + r + hf * 8) * D_MOD + n0 + j * 16 + nn] =
                acc[j][r] + bb;
    }
}

// ---------------------------------------------------------------------------
extern "C" void kernel_launch(void* const* d_in, const int* in_sizes, int n_in,
                              void* d_out, int out_size, void* d_ws,
                              size_t ws_size, hipStream_t stream) {
    const float* X     = (const float*)d_in[0];
    const float* Wq    = (const float*)d_in[1];
    const float* bq    = (const float*)d_in[2];
    const float* Wk    = (const float*)d_in[3];
    const float* bk    = (const float*)d_in[4];
    const float* Wv    = (const float*)d_in[5];
    const float* bv    = (const float*)d_in[6];
    const float* scale = (const float*)d_in[7];
    const float* Wo    = (const float*)d_in[8];
    const float* bo    = (const float*)d_in[9];
    float* out = (float*)d_out;

    unsigned short* ws = (unsigned short*)d_ws;
    size_t off = 0;
    unsigned short* Xb  = ws + off; off += (size_t)S_LEN * D_MOD;
    unsigned short* WqT = ws + off; off += (size_t)NHEAD * HDIM * D_MOD;
    unsigned short* WkT = ws + off; off += (size_t)NHEAD * HDIM * D_MOD;
    unsigned short* WvT = ws + off; off += (size_t)NHEAD * HDIM * D_MOD;
    unsigned short* WoT = ws + off; off += (size_t)D_MOD * D_MOD;
    unsigned short* Qb  = ws + off; off += (size_t)NHEAD * S_LEN * HDIM;
    unsigned short* Kb  = ws + off; off += (size_t)NHEAD * S_LEN * HDIM;
    unsigned short* Vt  = ws + off; off += (size_t)NHEAD * S_LEN * HDIM;
    unsigned short* Ob  = Xb;  // Xb dead after k_qkv; reuse for O

    int n = S_LEN * D_MOD;
    k_f32_to_bf16<<<(n + 255) / 256, 256, 0, stream>>>(X, Xb, n);
    n = NHEAD * HDIM * D_MOD;
    k_transpose_w<<<(n + 255) / 256, 256, 0, stream>>>(Wq, WqT);
    k_transpose_w<<<(n + 255) / 256, 256, 0, stream>>>(Wk, WkT);
    k_transpose_w<<<(n + 255) / 256, 256, 0, stream>>>(Wv, WvT);
    n = D_MOD * D_MOD;
    k_transpose_wo<<<(n + 255) / 256, 256, 0, stream>>>(Wo, WoT);

    k_qkv<<<dim3(S_LEN / 64, NHEAD, 3), 128, 0, stream>>>(
        Xb, WqT, WkT, WvT, bq, bk, bv, Qb, Kb, Vt);

    k_attn<<<dim3(S_LEN / 64, NHEAD), 128, 0, stream>>>(Qb, Kb, Vt, scale, Ob);

    k_out<<<dim3(S_LEN / 64, D_MOD / 64), 128, 0, stream>>>(Ob, WoT, bo, out);
}
